// Head_63797444215106
// MI455X (gfx1250) — matrix operand order repaired and checked by reading, not compiled, not run on hardware
//
#include <hip/hip_runtime.h>
#include <hip/hip_bf16.h>

typedef __bf16 bf16;
typedef __attribute__((ext_vector_type(16))) __bf16 v16bf;
typedef __attribute__((ext_vector_type(8)))  __bf16 v8bf;
typedef __attribute__((ext_vector_type(8)))  float  v8f;
typedef __attribute__((ext_vector_type(4)))  float  v4f;
typedef __attribute__((ext_vector_type(4)))  unsigned int v4u;
typedef __attribute__((ext_vector_type(8)))  int    v8i;
typedef __attribute__((ext_vector_type(4)))  int    v4i;

#define BATCH 16
#define SEQ   2048
#define NEMBD 1024
#define HID   64
#define HEAD  128

// workspace layout in bf16 elements
#define W1P_OFF 0                               // 3 * 65536 packed w1 frags
#define W2P_OFF (3 * 65536)                     // 3 * 8192  packed w2 frags
#define QB_OFF  262144                          // Q bf16 [B][T][H], pre-scaled
#define KB_OFF  (QB_OFF + BATCH * SEQ * HEAD)   // K bf16 [B][T][H]
#define VT_OFF  (KB_OFF + BATCH * SEQ * HEAD)   // V^T bf16 [B][H][T]

static __device__ __forceinline__ v8f vzero8() { v8f z = {}; return z; }

static __device__ __forceinline__ v8f wmma_bf16(v16bf a, v16bf b, v8f c) {
  // (neg_a, A, neg_b, B, c_mod, C, reuse_a, reuse_b)
  return __builtin_amdgcn_wmma_f32_16x16x32_bf16(false, a, false, b, (short)0, c,
                                                 false, false);
}

// 16 contiguous bf16 (32B) -> one fragment register set
static __device__ __forceinline__ v16bf ld_frag16(const bf16* p) {
  union { v16bf v; v8bf h[2]; } u;
  u.h[0] = *(const v8bf*)(p);
  u.h[1] = *(const v8bf*)(p + 8);
  return u.v;
}
// two separated 8-element groups (A-fragment pattern: K = kb..kb+7, kb+16..kb+23)
static __device__ __forceinline__ v16bf ld_frag8_8(const bf16* p0, const bf16* p1) {
  union { v16bf v; v8bf h[2]; } u;
  u.h[0] = *(const v8bf*)(p0);
  u.h[1] = *(const v8bf*)(p1);
  return u.v;
}
// fp32 source -> bf16 A-fragment (16 contiguous floats at +0)
static __device__ __forceinline__ v16bf ld_afrag_f32(const float* p) {
  v4f f0 = *(const v4f*)(p);
  v4f f1 = *(const v4f*)(p + 4);
  v4f f2 = *(const v4f*)(p + 8);
  v4f f3 = *(const v4f*)(p + 12);
  union { v16bf v; v8bf h[2]; } u;
#pragma unroll
  for (int i = 0; i < 4; ++i) {
    u.h[0][i]     = (bf16)f0[i];
    u.h[0][i + 4] = (bf16)f1[i];
    u.h[1][i]     = (bf16)f2[i];
    u.h[1][i + 4] = (bf16)f3[i];
  }
  return u.v;
}

// ---------------------------------------------------------------------------
// Tensor Data Mover: 2D tile load global -> LDS (bf16 elements).
// D# per CDNA5 ISA.  tile_d0 = contiguous run,
// tile_d1 = #lines, stride0 = line stride (elements).  Tracked by TENSORcnt.
// ---------------------------------------------------------------------------
static __device__ __forceinline__ void tdm_load_2d(unsigned int lds_addr,
                                                   const bf16* gptr,
                                                   unsigned int tile_d0,
                                                   unsigned int tile_d1,
                                                   unsigned int tensor_d0,
                                                   unsigned int tensor_d1,
                                                   unsigned long long stride0) {
  unsigned long long ga = (unsigned long long)(uintptr_t)gptr;
  v4u g0;
  g0[0] = 1u;                                   // count=1, user descriptor
  g0[1] = lds_addr;                             // LDS byte address
  g0[2] = (unsigned int)(ga & 0xffffffffull);   // global_addr[31:0]
  g0[3] = (unsigned int)((ga >> 32) & 0x01ffffffull) | (2u << 30); // type=2
  v8i g1;
  g1[0] = (int)(1u << 16);                      // data_size=1 (2 bytes)
  g1[1] = (int)((tensor_d0 & 0xffffu) << 16);   // tensor_dim0[15:0]
  g1[2] = (int)(((tensor_d0 >> 16) & 0xffffu) | ((tensor_d1 & 0xffffu) << 16));
  g1[3] = (int)(((tensor_d1 >> 16) & 0xffffu) | ((tile_d0 & 0xffffu) << 16));
  g1[4] = (int)(tile_d1 & 0xffffu);             // tile_dim1, tile_dim2=0
  g1[5] = (int)(unsigned int)(stride0 & 0xffffffffull);
  g1[6] = (int)(unsigned int)((stride0 >> 32) & 0xffffull);
  g1[7] = 0;
  v4i z = {0, 0, 0, 0};
#if __clang_major__ >= 23
  v8i z8 = {0, 0, 0, 0, 0, 0, 0, 0};
  __builtin_amdgcn_tensor_load_to_lds(g0, g1, z, z, z8, 0);
#else
  __builtin_amdgcn_tensor_load_to_lds(g0, g1, z, z, 0);
#endif
}

// ---------------------------------------------------------------------------
// Kernel 0: repack fp32 weights into bf16 B-fragment order.
// B-frag as packed here (contiguous k): lane n in [0,16): col N=n, K=0..15 ; lane n+16: K=16..31.
// ---------------------------------------------------------------------------
__global__ void pack_weights(const float* __restrict__ kw1, const float* __restrict__ kw2,
                             const float* __restrict__ qw1, const float* __restrict__ qw2,
                             const float* __restrict__ vw1, const float* __restrict__ vw2,
                             bf16* __restrict__ ws) {
  int idx = blockIdx.x * blockDim.x + threadIdx.x;
  const int W1TOT = 3 * 65536;
  const int W2TOT = 3 * 8192;
  if (idx < W1TOT) {
    int p  = idx >> 16;
    int r  = idx & 65535;
    int kk = r >> 11;
    int r2 = r & 2047;
    int nt = r2 >> 9;
    int r3 = r2 & 511;
    int lane = r3 >> 4;
    int e    = r3 & 15;
    int K = kk * 32 + ((lane >> 4) << 4) + e;
    int N = (nt << 4) + (lane & 15);
    const float* w = (p == 0) ? kw1 : (p == 1) ? qw1 : vw1;
    ws[W1P_OFF + idx] = (bf16)w[K * HID + N];
  } else if (idx < W1TOT + W2TOT) {
    int j  = idx - W1TOT;
    int p  = j >> 13;
    int r  = j & 8191;
    int kk = r >> 12;
    int r2 = r & 4095;
    int nt = r2 >> 9;
    int r3 = r2 & 511;
    int lane = r3 >> 4;
    int e    = r3 & 15;
    int K = kk * 32 + ((lane >> 4) << 4) + e;
    int N = (nt << 4) + (lane & 15);
    const float* w = (p == 0) ? kw2 : (p == 1) ? qw2 : vw2;
    ws[W2P_OFF + j] = (bf16)w[K * HEAD + N];
  }
}

// ---------------------------------------------------------------------------
// Kernel 1: fused QKV MLP.  One wave = 16 rows of x.
// Per-kstep w1 fragment set (3 proj x 2048 bf16, stride 65536) is TDM-staged
// into double-buffered LDS once per block; waves consume via ds_load_b128.
// ---------------------------------------------------------------------------
__global__ __launch_bounds__(256) void qkv_proj(
    const float* __restrict__ x,
    const float* __restrict__ kb1, const float* __restrict__ qb1,
    const float* __restrict__ vb1,
    bf16* __restrict__ ws) {
  __shared__ __align__(16) bf16 wbuf[2][3 * 2048];   // 2 x 12KB w1 fragments
  __shared__ __align__(16) bf16 hbuf[8][16 * HID];   // per-wave hidden staging
  const int lane = threadIdx.x & 31;
  const int w    = threadIdx.x >> 5;
  const int row0 = (blockIdx.x * 8 + w) * 16;
  const int lrow = lane & 15;
  const int hi   = lane >> 4;
  const int NK   = NEMBD / 32;

  v8f acc[3][4];
#pragma unroll
  for (int p = 0; p < 3; ++p)
#pragma unroll
    for (int nt = 0; nt < 4; ++nt) acc[p][nt] = vzero8();

  const bf16* w1p   = ws + W1P_OFF;
  const float* xrow = x + (size_t)(row0 + lrow) * NEMBD + hi * 16;

  // prologue: stage kstep 0 weight fragments
  if (w == 0) {
    tdm_load_2d((unsigned int)(uintptr_t)&wbuf[0][0], w1p,
                2048, 3, 2048, 3 * 32, 65536ull);
    __builtin_amdgcn_s_wait_tensorcnt(0);
  }
  __syncthreads();

  int cur = 0;
  for (int kk = 0; kk < NK; ++kk) {
    if (kk + 1 < NK && w == 0) {
      tdm_load_2d((unsigned int)(uintptr_t)&wbuf[cur ^ 1][0],
                  w1p + (kk + 1) * 2048, 2048, 3, 2048, 3 * 32, 65536ull);
    }
    __builtin_prefetch(xrow + (kk + 4) * 32, 0, 0);
    v16bf a = ld_afrag_f32(xrow + kk * 32);
#pragma unroll
    for (int p = 0; p < 3; ++p) {
#pragma unroll
      for (int nt = 0; nt < 4; ++nt) {
        const bf16* bp = &wbuf[cur][p * 2048 + ((nt * 32 + lane) << 4)];
        acc[p][nt] = wmma_bf16(a, ld_frag16(bp), acc[p][nt]);
      }
    }
    if (kk + 1 < NK) {
      if (w == 0) __builtin_amdgcn_s_wait_tensorcnt(0);
      __syncthreads();
      cur ^= 1;
    }
  }

  const bf16* w2p = ws + W2P_OFF;
  bf16* qb = ws + QB_OFF;
  bf16* kb = ws + KB_OFF;
  bf16* vt = ws + VT_OFF;
  const int batch = row0 / SEQ;
  const int t0    = row0 % SEQ;
  const float* b1s[3] = {kb1, qb1, vb1};
  const float qscale = 0.08838834764831845f;  // 1/sqrt(HEAD)

#pragma unroll
  for (int p = 0; p < 3; ++p) {
    float bias[4];
#pragma unroll
    for (int nt = 0; nt < 4; ++nt) bias[nt] = b1s[p][nt * 16 + lrow];
#pragma unroll
    for (int j = 0; j < 8; ++j) {
      int M = j + hi * 8;
#pragma unroll
      for (int nt = 0; nt < 4; ++nt) {
        float hv = tanhf(acc[p][nt][j] + bias[nt]);
        hbuf[w][M * HID + nt * 16 + lrow] = (bf16)hv;
      }
    }
    v16bf ha[2];
#pragma unroll
    for (int ks = 0; ks < 2; ++ks) {
      const bf16* hp = &hbuf[w][lrow * HID + ks * 32 + hi * 16];
      ha[ks] = ld_frag16(hp);
    }
    v8f o[8];
#pragma unroll
    for (int nt = 0; nt < 8; ++nt) o[nt] = vzero8();
#pragma unroll
    for (int ks = 0; ks < 2; ++ks)
#pragma unroll
      for (int nt = 0; nt < 8; ++nt) {
        const bf16* bp = w2p + p * 8192 + (((ks * 8 + nt) * 32 + lane) << 4);
        o[nt] = wmma_bf16(ha[ks], ld_frag16(bp), o[nt]);
      }
#pragma unroll
    for (int j = 0; j < 8; ++j) {
      int M = j + hi * 8;
#pragma unroll
      for (int nt = 0; nt < 8; ++nt) {
        int N = nt * 16 + lrow;
        float val = o[nt][j];
        if (p == 0) {
          kb[(size_t)(row0 + M) * HEAD + N] = (bf16)val;
        } else if (p == 1) {
          qb[(size_t)(row0 + M) * HEAD + N] = (bf16)(val * qscale);
        } else {
          vt[((size_t)batch * HEAD + N) * SEQ + (t0 + M)] = (bf16)val;
        }
      }
    }
  }
}

// ---------------------------------------------------------------------------
// Kernel 2: causal flash attention, block-cooperative K/V staging via TDM.
// Block = 8 waves = 128 consecutive queries; double-buffered 64-key tiles.
// ---------------------------------------------------------------------------
#define KBLK 64

__global__ __launch_bounds__(256) void attn(
    const bf16* __restrict__ ws_q, const bf16* __restrict__ ws_k,
    const bf16* __restrict__ ws_vt, float* __restrict__ out) {
  __shared__ __align__(16) bf16 kbuf[2][KBLK * HEAD];   // 2 x 16KB K rows
  __shared__ __align__(16) bf16 vbuf[2][HEAD * KBLK];   // 2 x 16KB V^T rows
  __shared__ __align__(16) bf16 pbuf[8][16 * 32];       // per-wave P staging
  const int lane = threadIdx.x & 31;
  const int w    = threadIdx.x >> 5;
  const int b    = blockIdx.x >> 4;
  const int qbase = (blockIdx.x & 15) * 128;
  const int q0   = qbase + w * 16;
  const int lrow = lane & 15;
  const int hi   = lane >> 4;

  const bf16* qb = ws_q  + (size_t)b * SEQ * HEAD;
  const bf16* kb = ws_k  + (size_t)b * SEQ * HEAD;
  const bf16* vt = ws_vt + (size_t)b * HEAD * SEQ;

  // Q tile resident: 4 A-fragments over H=128 (already scaled by 1/sqrt(H))
  v16bf qa[4];
#pragma unroll
  for (int ks = 0; ks < 4; ++ks) {
    const bf16* qp = qb + (size_t)(q0 + lrow) * HEAD + ks * 32 + hi * 16;
    qa[ks] = ld_frag16(qp);
  }

  v8f o[8];
#pragma unroll
  for (int nt = 0; nt < 8; ++nt) o[nt] = vzero8();
  float m[8], s[8];
#pragma unroll
  for (int j = 0; j < 8; ++j) { m[j] = -1e30f; s[j] = 0.0f; }

  const int qmax = qbase + 112;          // last wave's q0
  const int nkb  = qmax / KBLK + 1;      // number of 64-key blocks needed

  // prologue: stage block 0 (wave0 -> K tile, wave1 -> V tile)
  if (w == 0) {
    tdm_load_2d((unsigned int)(uintptr_t)&kbuf[0][0], kb, HEAD, KBLK,
                HEAD, SEQ, HEAD);
    __builtin_amdgcn_s_wait_tensorcnt(0);
  } else if (w == 1) {
    tdm_load_2d((unsigned int)(uintptr_t)&vbuf[0][0], vt, KBLK, HEAD,
                SEQ, HEAD, SEQ);
    __builtin_amdgcn_s_wait_tensorcnt(0);
  }
  __syncthreads();

  int cur = 0;
  for (int ib = 0; ib < nkb; ++ib) {
    const int kt = ib * KBLK;
    // stage next block into the alternate buffer while computing this one
    if (ib + 1 < nkb) {
      if (w == 0) {
        tdm_load_2d((unsigned int)(uintptr_t)&kbuf[cur ^ 1][0],
                    kb + (size_t)(kt + KBLK) * HEAD, HEAD, KBLK, HEAD, SEQ, HEAD);
      } else if (w == 1) {
        tdm_load_2d((unsigned int)(uintptr_t)&vbuf[cur ^ 1][0],
                    vt + (kt + KBLK), KBLK, HEAD, SEQ, HEAD, SEQ);
      }
    }

#pragma unroll
    for (int sub = 0; sub < 2; ++sub) {
      const int kt32 = kt + sub * 32;
      if (kt32 <= q0) {                 // wave-uniform causal block skip
        v8f af0 = vzero8(), af1 = vzero8();
#pragma unroll
        for (int ks = 0; ks < 4; ++ks) {
          const bf16* kp0 = &kbuf[cur][(sub * 32 + lrow) * HEAD + ks * 32 + hi * 16];
          const bf16* kp1 = kp0 + 16 * HEAD;
          af0 = wmma_bf16(qa[ks], ld_frag16(kp0), af0);
          af1 = wmma_bf16(qa[ks], ld_frag16(kp1), af1);
        }

        const bool edge = (kt32 + 32 > q0);
        float sc[8];
#pragma unroll
        for (int j = 0; j < 8; ++j) {
          int qg = q0 + j + hi * 8;
          float a0 = af0[j], a1 = af1[j];
          if (edge) {
            if (kt32 + lrow > qg)      a0 = -1e30f;
            if (kt32 + 16 + lrow > qg) a1 = -1e30f;
          }
          float rm = fmaxf(a0, a1);
#pragma unroll
          for (int d = 1; d < 16; d <<= 1) rm = fmaxf(rm, __shfl_xor(rm, d, 16));
          float mn  = fmaxf(m[j], rm);
          float scj = __expf(m[j] - mn);
          m[j] = mn;
          float p0 = __expf(a0 - mn);
          float p1 = __expf(a1 - mn);
          float rs = p0 + p1;
#pragma unroll
          for (int d = 1; d < 16; d <<= 1) rs += __shfl_xor(rs, d, 16);
          s[j]  = s[j] * scj + rs;
          sc[j] = scj;
          int M = j + hi * 8;
          pbuf[w][M * 32 + lrow]      = (bf16)p0;
          pbuf[w][M * 32 + 16 + lrow] = (bf16)p1;
        }
#pragma unroll
        for (int nt = 0; nt < 8; ++nt)
#pragma unroll
          for (int j = 0; j < 8; ++j) o[nt][j] *= sc[j];

        const bf16* pp = &pbuf[w][lrow * 32 + hi * 16];
        v16bf pa = ld_frag16(pp);
#pragma unroll
        for (int nt = 0; nt < 8; ++nt) {
          const bf16* vp = &vbuf[cur][(nt * 16 + lrow) * KBLK + sub * 32 + hi * 16];
          o[nt] = wmma_bf16(pa, ld_frag16(vp), o[nt]);
        }
      }
    }

    if (ib + 1 < nkb) {
      if (w <= 1) __builtin_amdgcn_s_wait_tensorcnt(0);
      __syncthreads();                  // next buffer ready, this one fully read
      cur ^= 1;
    }
  }

  // normalize and write fp32 output
#pragma unroll
  for (int j = 0; j < 8; ++j) {
    float inv = 1.0f / s[j];
    int row = q0 + j + hi * 8;
#pragma unroll
    for (int nt = 0; nt < 8; ++nt)
      out[((size_t)b * SEQ + row) * HEAD + nt * 16 + lrow] = o[nt][j] * inv;
  }
}

extern "C" void kernel_launch(void* const* d_in, const int* in_sizes, int n_in,
                              void* d_out, int out_size, void* d_ws, size_t ws_size,
                              hipStream_t stream) {
  const float* x    = (const float*)d_in[0];
  const float* kw1  = (const float*)d_in[1];
  const float* kb1v = (const float*)d_in[2];
  const float* kw2  = (const float*)d_in[3];
  const float* qw1  = (const float*)d_in[4];
  const float* qb1v = (const float*)d_in[5];
  const float* qw2  = (const float*)d_in[6];
  const float* vw1  = (const float*)d_in[7];
  const float* vb1v = (const float*)d_in[8];
  const float* vw2  = (const float*)d_in[9];
  bf16* ws   = (bf16*)d_ws;
  float* out = (float*)d_out;

  const int packTotal = 3 * 65536 + 3 * 8192;
  pack_weights<<<(packTotal + 255) / 256, 256, 0, stream>>>(kw1, kw2, qw1, qw2,
                                                            vw1, vw2, ws);
  qkv_proj<<<256, 256, 0, stream>>>(x, kb1v, qb1v, vb1v, ws);
  attn<<<256, 256, 0, stream>>>(ws + QB_OFF, ws + KB_OFF, ws + VT_OFF, out);
}
